// MoeBlock_33225867002145
// MI455X (gfx1250) — matrix operand order repaired and checked by reading, not compiled, not run on hardware
//
#include <hip/hip_runtime.h>
#include <cstdint>

// ---------------- types ----------------
typedef __attribute__((ext_vector_type(16))) __bf16 v16bf;
typedef __attribute__((ext_vector_type(8)))  __bf16 v8bf;
typedef __attribute__((ext_vector_type(8)))  float  v8f;
typedef int v4i32 __attribute__((vector_size(16)));

// ---------------- problem constants ----------------
#define CB   2
#define CT   1024
#define CC   1024
#define CH   16
#define CD   64
#define CNH  4096
#define CE   2
#define CWIN 256
#define NTOK (CB*CT)   // 2048
#define HB   (CH*CB)   // 32
#define QKVN (3*CC)    // 3072 = qsum | k | v

#ifndef __has_builtin
#define __has_builtin(x) 0
#endif

#if __has_builtin(__builtin_amdgcn_global_load_async_to_lds_b128)
#define HAS_ASYNC_LDS 1
#endif

#if __has_builtin(__builtin_amdgcn_s_wait_asynccnt)
#define WAIT_ASYNCCNT() __builtin_amdgcn_s_wait_asynccnt(0)
#else
#define WAIT_ASYNCCNT() asm volatile("s_wait_asynccnt 0" ::: "memory")
#endif

// ---------------- generic bf16 WMMA GEMM ----------------
// C[M,N] = A[M,K] * B[K,N] (+bias +res), A bf16 K-contig, B given transposed
// (mode 0) as Bt[N][K] K-contig, or gathered (modes 1/2).
// Double-buffered LDS: async DMA of tile i+1 overlaps WMMA on tile i.
struct GemmP {
  const __bf16* A;  long lda;
  const __bf16* Bt; long ldb;
  float* outF; __bf16* outBF; long ldc;
  const float* bias;
  const float* res;
  int M, N, K;
  int bmode;   // 0 = Bt contiguous, 1 = KR reinterpret gather, 2 = row-major gather
  int bz0;     // z = z1*bz0 + z0  (>=1)
  long sA0, sA1, sB0, sB1, sC0, sC1;
};

#define BM 128
#define BN 128
#define BK 32

__device__ __forceinline__ void stage16B(const __bf16* g, __bf16* l) {
#ifdef HAS_ASYNC_LDS
  __builtin_amdgcn_global_load_async_to_lds_b128(
      (v4i32 __attribute__((address_space(1)))*)(uintptr_t)(const void*)g,
      (v4i32 __attribute__((address_space(3)))*)(uint32_t)(uintptr_t)(void*)l,
      0, 0);
#else
  *(v8bf*)l = *(const v8bf*)g;
#endif
}

__launch_bounds__(256)
__global__ void gemm_bf16_wmma(GemmP p) {
  __shared__ __align__(16) __bf16 As[2][BM][BK];
  __shared__ __align__(16) __bf16 Bs[2][BN][BK];

  const int tid  = threadIdx.x;
  const int lane = tid & 31;
  const int wid  = tid >> 5;
  const int l16  = lane & 15;
  const int half = lane >> 4;
  const int wm   = wid >> 2;   // 0..1
  const int wn   = wid & 3;    // 0..3
  const int row0 = blockIdx.y * BM;
  const int col0 = blockIdx.x * BN;

  const int z  = blockIdx.z;
  const int z0 = z % p.bz0;
  const int z1 = z / p.bz0;
  const __bf16* Ab = p.A  + (long)z0 * p.sA0 + (long)z1 * p.sA1;
  const __bf16* Bb = p.Bt + (long)z0 * p.sB0 + (long)z1 * p.sB1;
  const long    zC = (long)z0 * p.sC0 + (long)z1 * p.sC1;

  auto stage_tile = [&](int buf, int k0) {
    // ---- A tile 128x32 bf16: 512 x 16B chunks, 2 per thread ----
    for (int i = 0; i < 2; ++i) {
      int cid = tid + i * 256;
      int r   = cid >> 2;
      int kk  = (cid & 3) << 3;
      stage16B(Ab + (long)(row0 + r) * p.lda + (k0 + kk), &As[buf][r][kk]);
    }
    // ---- B tile as Bs[n][k] ----
    if (p.bmode == 0) {
      for (int i = 0; i < 2; ++i) {
        int cid = tid + i * 256;
        int n   = cid >> 2;
        int kk  = (cid & 3) << 3;
        stage16B(Bb + (long)(col0 + n) * p.ldb + (k0 + kk), &Bs[buf][n][kk]);
      }
    } else if (p.bmode == 1) {
      // kr[d][s] = kv[(d*16 + s/64)*ldb + (s&63)]  (k.reshape(H,B,D,T) bug-faithful)
      for (int i = 0; i < 16; ++i) {
        int e = tid * 16 + i;
        int n = e >> 5, k = e & 31;
        int s = col0 + n, d = k0 + k;
        Bs[buf][n][k] = Bb[(long)(d * 16 + (s >> 6)) * p.ldb + (s & 63)];
      }
    } else {
      // plain row-major B[k][n]
      for (int i = 0; i < 16; ++i) {
        int e = tid * 16 + i;
        int n = e >> 5, k = e & 31;
        int col = col0 + n;
        Bs[buf][n][k] = (col < p.N) ? Bb[(long)(k0 + k) * p.ldb + col] : (__bf16)0.f;
      }
    }
  };

  const v8f zacc = {0.f,0.f,0.f,0.f,0.f,0.f,0.f,0.f};
  v8f acc[4][2];
  for (int i = 0; i < 4; ++i)
    for (int j = 0; j < 2; ++j) acc[i][j] = zacc;

  const int nk = p.K / BK;
  stage_tile(0, 0);

  for (int it = 0; it < nk; ++it) {
    const int cur = it & 1;
#ifdef HAS_ASYNC_LDS
    WAIT_ASYNCCNT();                 // tile `it` landed in LDS (per-wave)
#endif
    __syncthreads();                 // all waves staged tile `it`; buf `cur^1` free

    if (it + 1 < nk) stage_tile(cur ^ 1, (it + 1) * BK);   // DMA overlaps WMMAs
    if (it + 2 < nk) {
      __builtin_prefetch(Ab + (long)(row0 + (tid >> 1)) * p.lda + (it + 2) * BK, 0, 1);
      if (p.bmode == 0)
        __builtin_prefetch(Bb + (long)(col0 + (tid >> 1)) * p.ldb + (it + 2) * BK, 0, 1);
    }

    // ---- fragments per ISA 16-bit A 16x32 / B layouts ----
    v16bf af[4], bfr[2];
    for (int mt = 0; mt < 4; ++mt) {
      int m = wm * 64 + mt * 16 + l16;
      v8bf lo = *(const v8bf*)&As[cur][m][half * 8];
      v8bf hi = *(const v8bf*)&As[cur][m][16 + half * 8];
      af[mt] = __builtin_shufflevector(lo, hi, 0,1,2,3,4,5,6,7,8,9,10,11,12,13,14,15);
    }
    for (int nt = 0; nt < 2; ++nt) {
      int n = wn * 32 + nt * 16 + l16;
      v8bf lo = *(const v8bf*)&Bs[cur][n][half * 8];
      v8bf hi = *(const v8bf*)&Bs[cur][n][16 + half * 8];
      bfr[nt] = __builtin_shufflevector(lo, hi, 0,1,2,3,4,5,6,7,8,9,10,11,12,13,14,15);
    }
    for (int mt = 0; mt < 4; ++mt)
      for (int nt = 0; nt < 2; ++nt)
        acc[mt][nt] = __builtin_amdgcn_wmma_f32_16x16x32_bf16(
            false, af[mt], false, bfr[nt], (short)0, acc[mt][nt], false, false);
  }

  // ---- epilogue: C/D layout lane n = l16, m = half*8 + r ----
  for (int nt = 0; nt < 2; ++nt) {
    int col = col0 + wn * 32 + nt * 16 + l16;
    if (col >= p.N) continue;
    float bias = p.bias ? p.bias[col] : 0.f;
    for (int mt = 0; mt < 4; ++mt) {
      int mb = row0 + wm * 64 + mt * 16 + half * 8;
      for (int r = 0; r < 8; ++r) {
        int row = mb + r;
        if (row >= p.M) continue;
        long off = zC + (long)row * p.ldc + col;
        float v = acc[mt][nt][r] + bias;
        if (p.res)   v += p.res[off];
        if (p.outF)  __builtin_nontemporal_store(v, &p.outF[off]);
        if (p.outBF) __builtin_nontemporal_store((__bf16)v, &p.outBF[off]);
      }
    }
  }
}

// ---------------- weight convert + transpose: Wt[N][K] = bf16(W[K][N]) ----------------
__launch_bounds__(256)
__global__ void cvt_t_kernel(const float* __restrict__ W, __bf16* __restrict__ Wt,
                             int K, int N) {
  __shared__ float tile[32][33];
  int nb = blockIdx.x * 32, kb = blockIdx.y * 32;
  int tx = threadIdx.x & 31, ty = threadIdx.x >> 5;
  for (int i = 0; i < 4; ++i) {
    int k = ty + i * 8;
    tile[k][tx] = ((kb + k) < K && (nb + tx) < N) ? W[(long)(kb + k) * N + nb + tx] : 0.f;
  }
  __syncthreads();
  for (int i = 0; i < 4; ++i) {
    int n = ty + i * 8;
    if ((nb + n) < N && (kb + tx) < K)
      Wt[(long)(nb + n) * K + kb + tx] = (__bf16)tile[tx][n];
  }
}

// Build transposed bf16 QKV weight [3072][1024]; q block pre-sums the Q group.
__global__ void build_wqkv_kernel(const float* __restrict__ Wq, const float* __restrict__ Wk,
                                  const float* __restrict__ Wv, __bf16* __restrict__ Wt) {
  long i = (long)blockIdx.x * 256 + threadIdx.x;   // QKVN*CC threads
  int n = (int)(i >> 10);
  int c = (int)(i & (CC - 1));
  float v;
  if (n < CC) {                       // q-sum
    int h = n >> 6, d = n & 63;
    const float* base = Wq + (long)h * 2 * CC * CD;
    v = base[(long)c * CD + d] + base[(long)CC * CD + (long)c * CD + d];
  } else if (n < 2 * CC) {            // k
    int nn = n - CC; int h = nn >> 6, d = nn & 63;
    v = Wk[((long)h * CC + c) * CD + d];
  } else {                            // v
    int nn = n - 2 * CC; int h = nn >> 6, d = nn & 63;
    v = Wv[((long)h * CC + c) * CD + d];
  }
  Wt[(long)n * CC + c] = (__bf16)v;
}

// ---------------- layernorm (unbiased var, like torch .var()) ----------------
__launch_bounds__(256)
__global__ void ln_kernel(const float* __restrict__ x, const float* __restrict__ g,
                          const float* __restrict__ b, float* outF, __bf16* outBF) {
  __shared__ float red[256];
  const int row = blockIdx.x;
  const float* xr = x + (long)row * CC;
  float s = 0.f, ss = 0.f;
  for (int c = threadIdx.x; c < CC; c += 256) { float v = xr[c]; s += v; ss += v * v; }
  red[threadIdx.x] = s; __syncthreads();
  for (int o = 128; o > 0; o >>= 1) { if (threadIdx.x < o) red[threadIdx.x] += red[threadIdx.x + o]; __syncthreads(); }
  float S = red[0]; __syncthreads();
  red[threadIdx.x] = ss; __syncthreads();
  for (int o = 128; o > 0; o >>= 1) { if (threadIdx.x < o) red[threadIdx.x] += red[threadIdx.x + o]; __syncthreads(); }
  float SS = red[0];
  float mean = S / CC;
  float inv  = rsqrtf((SS - S * mean) / (CC - 1) + 1e-5f);
  for (int c = threadIdx.x; c < CC; c += 256) {
    float y = (xr[c] - mean) * inv * g[c] + b[c];
    long off = (long)row * CC + c;
    if (outF) outF[off] = y;
    outBF[off] = (__bf16)y;
  }
}

// ---------------- scale + window mask + softmax, f32 -> bf16 ----------------
__launch_bounds__(256)
__global__ void softmax_mask_kernel(const float* __restrict__ scores, __bf16* __restrict__ w) {
  __shared__ float red[256];
  const int t  = blockIdx.x;
  const int hb = blockIdx.y;
  const long base = ((long)hb * CT + t) * CT;
  const float scale = 0.125f;          // D^-0.5
  float v[4];
  float mx = -INFINITY;
  for (int i = 0; i < 4; ++i) {
    int s = threadIdx.x + i * 256;
    float xv = scores[base + s] * scale;
    if (s + CWIN <= t) xv = -INFINITY; // tril(-WINDOW) past mask only (bug-faithful)
    v[i] = xv; mx = fmaxf(mx, xv);
  }
  red[threadIdx.x] = mx; __syncthreads();
  for (int o = 128; o > 0; o >>= 1) { if (threadIdx.x < o) red[threadIdx.x] = fmaxf(red[threadIdx.x], red[threadIdx.x + o]); __syncthreads(); }
  mx = red[0]; __syncthreads();
  float sum = 0.f;
  for (int i = 0; i < 4; ++i) { v[i] = __expf(v[i] - mx); sum += v[i]; }
  red[threadIdx.x] = sum; __syncthreads();
  for (int o = 128; o > 0; o >>= 1) { if (threadIdx.x < o) red[threadIdx.x] += red[threadIdx.x + o]; __syncthreads(); }
  float inv = 1.f / red[0];
  for (int i = 0; i < 4; ++i) {
    int s = threadIdx.x + i * 256;
    w[base + s] = (__bf16)(v[i] * inv);
  }
}

// ---------------- router: logits, top-1 ----------------
__launch_bounds__(256)
__global__ void router_kernel(const float* __restrict__ xn2, const float* __restrict__ Wg,
                              const float* __restrict__ bg, float* topv, int* topi) {
  __shared__ float r0[256], r1[256];
  int row = blockIdx.x;
  const float* xr = xn2 + (long)row * CC;
  float a0 = 0.f, a1 = 0.f;
  for (int c = threadIdx.x; c < CC; c += 256) {
    float v = xr[c]; a0 += v * Wg[c * 2 + 0]; a1 += v * Wg[c * 2 + 1];
  }
  r0[threadIdx.x] = a0; r1[threadIdx.x] = a1; __syncthreads();
  for (int o = 128; o > 0; o >>= 1) {
    if (threadIdx.x < o) { r0[threadIdx.x] += r0[threadIdx.x + o]; r1[threadIdx.x] += r1[threadIdx.x + o]; }
    __syncthreads();
  }
  if (threadIdx.x == 0) {
    float l0 = r0[0] + bg[0], l1 = r1[0] + bg[1];
    int e = (l1 > l0) ? 1 : 0;
    topi[row] = e;
    topv[row] = e ? l1 : l0;
  }
}

// per-batch softmax stats over tokens (faithful softmax over token dim)
__launch_bounds__(256)
__global__ void bstats_kernel(const float* __restrict__ topv, float* bstats) {
  __shared__ float red[256];
  int b = blockIdx.x;
  const float* tv = topv + (long)b * CT;
  float mx = -INFINITY;
  for (int t = threadIdx.x; t < CT; t += 256) mx = fmaxf(mx, tv[t]);
  red[threadIdx.x] = mx; __syncthreads();
  for (int o = 128; o > 0; o >>= 1) { if (threadIdx.x < o) red[threadIdx.x] = fmaxf(red[threadIdx.x], red[threadIdx.x + o]); __syncthreads(); }
  float m = red[0]; __syncthreads();
  float s = 0.f;
  for (int t = threadIdx.x; t < CT; t += 256) s += __expf(tv[t] - m);
  red[threadIdx.x] = s; __syncthreads();
  for (int o = 128; o > 0; o >>= 1) { if (threadIdx.x < o) red[threadIdx.x] += red[threadIdx.x + o]; __syncthreads(); }
  if (threadIdx.x == 0) { bstats[b * 2] = m; bstats[b * 2 + 1] = red[0]; }
}

// ---------------- swiglu: act = swish_beta(sx) * glu -> bf16 ----------------
__global__ void swiglu_kernel(const float* __restrict__ sx, const float* __restrict__ glu,
                              const float* __restrict__ Ebeta, int e, __bf16* __restrict__ act) {
  long i = (long)blockIdx.x * 256 + threadIdx.x;
  float beta = Ebeta[e];
  float s = sx[i], g = glu[i];
  float sig = 1.f / (1.f + __expf(-beta * s));
  act[i] = (__bf16)(s * sig * g);
}

// ---------------- final mix: out = x1 + coef[topi] * eout[topi] ----------------
__global__ void mix_kernel(const float* __restrict__ x1, const float* __restrict__ eout,
                           const float* __restrict__ topv, const int* __restrict__ topi,
                           const float* __restrict__ bstats, float* __restrict__ out) {
  long i = (long)blockIdx.x * 256 + threadIdx.x;
  int row = (int)(i >> 10);     // / C
  int b   = row >> 10;          // / T
  float m = bstats[b * 2], ssum = bstats[b * 2 + 1];
  float wgt = __expf(topv[row] - m) / ssum;
  int e = topi[row];
  out[i] = x1[i] + wgt * eout[(long)e * NTOK * CC + i];
}

// ---------------- host orchestration ----------------
extern "C" void kernel_launch(void* const* d_in, const int* in_sizes, int n_in,
                              void* d_out, int out_size, void* d_ws, size_t ws_size,
                              hipStream_t stream) {
  (void)in_sizes; (void)n_in; (void)out_size; (void)ws_size;
  const float* x     = (const float*)d_in[0];
  const float* Wq    = (const float*)d_in[1];
  const float* Wk    = (const float*)d_in[2];
  const float* Wv    = (const float*)d_in[3];
  const float* Wp    = (const float*)d_in[4];
  const float* bp    = (const float*)d_in[5];
  const float* ln1g  = (const float*)d_in[6];
  const float* ln1b  = (const float*)d_in[7];
  const float* ln2g  = (const float*)d_in[8];
  const float* ln2b  = (const float*)d_in[9];
  const float* Wg    = (const float*)d_in[10];
  const float* bg    = (const float*)d_in[11];
  const float* Ew1   = (const float*)d_in[12];
  const float* Eb1   = (const float*)d_in[13];
  const float* Ews   = (const float*)d_in[14];
  const float* Ebs   = (const float*)d_in[15];
  const float* Ebeta = (const float*)d_in[16];
  const float* Ewg   = (const float*)d_in[17];
  const float* Ebg   = (const float*)d_in[18];
  const float* Ew2   = (const float*)d_in[19];
  const float* Eb2   = (const float*)d_in[20];
  float* out = (float*)d_out;

  char* wsp = (char*)d_ws;
  auto carve = [&](size_t bytes) -> char* {
    char* p = wsp; wsp += (bytes + 255) & ~(size_t)255; return p;
  };
  __bf16* wqkvt  = (__bf16*)carve((size_t)QKVN * CC * 2);
  __bf16* wpt    = (__bf16*)carve((size_t)CC * CC * 2);
  __bf16* ew1t   = (__bf16*)carve((size_t)CE * CNH * CC * 2);
  __bf16* ewst   = (__bf16*)carve((size_t)CE * CNH * CNH * 2);
  __bf16* ewgt   = (__bf16*)carve((size_t)CE * CNH * CNH * 2);
  __bf16* ew2t   = (__bf16*)carve((size_t)CE * CC * CNH * 2);
  __bf16* xn1bf  = (__bf16*)carve((size_t)NTOK * CC * 2);
  __bf16* qkvbf  = (__bf16*)carve((size_t)NTOK * QKVN * 2);
  float*  scoresf= (float*)carve((size_t)HB * CT * CT * 4);
  __bf16* wbf    = (__bf16*)carve((size_t)HB * CT * CT * 2);
  __bf16* obf    = (__bf16*)carve((size_t)NTOK * CC * 2);
  float*  x1f    = (float*)carve((size_t)NTOK * CC * 4);
  float*  xn2f   = (float*)carve((size_t)NTOK * CC * 4);
  __bf16* xn2bf  = (__bf16*)carve((size_t)NTOK * CC * 2);
  float*  topv   = (float*)carve((size_t)NTOK * 4);
  int*    topi   = (int*)carve((size_t)NTOK * 4);
  float*  bstats = (float*)carve((size_t)CB * 2 * 4);
  __bf16* h1bf   = (__bf16*)carve((size_t)NTOK * CNH * 2);
  float*  sxf    = (float*)carve((size_t)NTOK * CNH * 4);
  float*  gluf   = (float*)carve((size_t)NTOK * CNH * 4);
  __bf16* actbf  = (__bf16*)carve((size_t)NTOK * CNH * 2);
  float*  eoutf  = (float*)carve((size_t)CE * NTOK * CC * 4);

  auto launch_gemm = [&](GemmP& p, int nz) {
    dim3 grid((p.N + BN - 1) / BN, (p.M + BM - 1) / BM, nz);
    gemm_bf16_wmma<<<grid, 256, 0, stream>>>(p);
  };

  // ---- weight convert/transpose to bf16 ----
  build_wqkv_kernel<<<(QKVN * CC) / 256, 256, 0, stream>>>(Wq, Wk, Wv, wqkvt);
  cvt_t_kernel<<<dim3(CC / 32, CC / 32), 256, 0, stream>>>(Wp, wpt, CC, CC);
  for (int e = 0; e < CE; ++e) {
    cvt_t_kernel<<<dim3(CNH / 32, CC / 32), 256, 0, stream>>>(Ew1 + (long)e * CC * CNH,  ew1t + (long)e * CNH * CC, CC, CNH);
    cvt_t_kernel<<<dim3(CNH / 32, CNH / 32), 256, 0, stream>>>(Ews + (long)e * CNH * CNH, ewst + (long)e * CNH * CNH, CNH, CNH);
    cvt_t_kernel<<<dim3(CNH / 32, CNH / 32), 256, 0, stream>>>(Ewg + (long)e * CNH * CNH, ewgt + (long)e * CNH * CNH, CNH, CNH);
    cvt_t_kernel<<<dim3(CC / 32, CNH / 32), 256, 0, stream>>>(Ew2 + (long)e * CNH * CC,  ew2t + (long)e * CC * CNH, CNH, CC);
  }

  // ---- attention ----
  ln_kernel<<<NTOK, 256, 0, stream>>>(x, ln1g, ln1b, nullptr, xn1bf);

  GemmP p{};
  // qkv: [2048 x 1024] @ [1024 x 3072]
  p = GemmP{}; p.A = xn1bf; p.lda = CC; p.Bt = wqkvt; p.ldb = CC;
  p.outBF = qkvbf; p.ldc = QKVN; p.M = NTOK; p.N = QKVN; p.K = CC;
  p.bmode = 0; p.bz0 = 1;
  launch_gemm(p, 1);

  // scores: per (h,b) [T x D] @ kr[D x T] (reshape-gather), z0=b z1=h
  p = GemmP{}; p.A = qkvbf; p.lda = QKVN; p.Bt = qkvbf + CC; p.ldb = QKVN;
  p.outF = scoresf; p.ldc = CT; p.M = CT; p.N = CT; p.K = CD;
  p.bmode = 1; p.bz0 = CB;
  p.sA0 = (long)CT * QKVN; p.sA1 = CD;
  p.sB0 = (long)CT * QKVN; p.sB1 = CD;
  p.sC0 = (long)CT * CT;   p.sC1 = 2L * CT * CT;
  launch_gemm(p, HB);

  softmax_mask_kernel<<<dim3(CT, HB), 256, 0, stream>>>(scoresf, wbf);

  // o = w @ v: per (h,b) [T x T] @ [T x D], v gathered row-major
  p = GemmP{}; p.A = wbf; p.lda = CT; p.Bt = qkvbf + 2 * CC; p.ldb = QKVN;
  p.outBF = obf; p.ldc = CC; p.M = CT; p.N = CD; p.K = CT;
  p.bmode = 2; p.bz0 = CB;
  p.sA0 = (long)CT * CT;   p.sA1 = 2L * CT * CT;
  p.sB0 = (long)CT * QKVN; p.sB1 = CD;
  p.sC0 = (long)CT * CC;   p.sC1 = CD;
  launch_gemm(p, HB);

  // proj + residual: x1 = x + o @ Wp + bp
  p = GemmP{}; p.A = obf; p.lda = CC; p.Bt = wpt; p.ldb = CC;
  p.outF = x1f; p.ldc = CC; p.bias = bp; p.res = x;
  p.M = NTOK; p.N = CC; p.K = CC; p.bmode = 0; p.bz0 = 1;
  launch_gemm(p, 1);

  // ---- MoE ----
  ln_kernel<<<NTOK, 256, 0, stream>>>(x1f, ln2g, ln2b, xn2f, xn2bf);
  router_kernel<<<NTOK, 256, 0, stream>>>(xn2f, Wg, bg, topv, topi);
  bstats_kernel<<<CB, 256, 0, stream>>>(topv, bstats);

  for (int e = 0; e < CE; ++e) {
    // h1 = xn2 @ Ew1 + Eb1
    p = GemmP{}; p.A = xn2bf; p.lda = CC; p.Bt = ew1t + (long)e * CNH * CC; p.ldb = CC;
    p.outBF = h1bf; p.ldc = CNH; p.bias = Eb1 + (long)e * CNH;
    p.M = NTOK; p.N = CNH; p.K = CC; p.bmode = 0; p.bz0 = 1;
    launch_gemm(p, 1);
    // sx = h1 @ Ews + Ebs
    p = GemmP{}; p.A = h1bf; p.lda = CNH; p.Bt = ewst + (long)e * CNH * CNH; p.ldb = CNH;
    p.outF = sxf; p.ldc = CNH; p.bias = Ebs + (long)e * CNH;
    p.M = NTOK; p.N = CNH; p.K = CNH; p.bmode = 0; p.bz0 = 1;
    launch_gemm(p, 1);
    // glu = h1 @ Ewg + Ebg
    p = GemmP{}; p.A = h1bf; p.lda = CNH; p.Bt = ewgt + (long)e * CNH * CNH; p.ldb = CNH;
    p.outF = gluf; p.ldc = CNH; p.bias = Ebg + (long)e * CNH;
    p.M = NTOK; p.N = CNH; p.K = CNH; p.bmode = 0; p.bz0 = 1;
    launch_gemm(p, 1);

    swiglu_kernel<<<((long)NTOK * CNH) / 256, 256, 0, stream>>>(sxf, gluf, Ebeta, e, actbf);

    // eout = act @ Ew2 + Eb2
    p = GemmP{}; p.A = actbf; p.lda = CNH; p.Bt = ew2t + (long)e * CC * CNH; p.ldb = CNH;
    p.outF = eoutf + (long)e * NTOK * CC; p.ldc = CC; p.bias = Eb2 + (long)e * CC;
    p.M = NTOK; p.N = CC; p.K = CNH; p.bmode = 0; p.bz0 = 1;
    launch_gemm(p, 1);
  }

  mix_kernel<<<((long)NTOK * CC) / 256, 256, 0, stream>>>(x1f, eoutf, topv, topi, bstats, out);
}
